// Online_CP_multi_net_33380485824584
// MI455X (gfx1250) — compile-verified, matrix operand order checked by reading, not run on hardware
//
#include <hip/hip_runtime.h>
#include <hip/hip_bf16.h>

typedef __attribute__((ext_vector_type(8)))  _Float16 v8h;
typedef __attribute__((ext_vector_type(16))) _Float16 v16h;
typedef __attribute__((ext_vector_type(8)))  float    v8f;

#define OMEGA 1.5f
#define MID   256
#define RPAD  128   // R=100 zero-padded to 128 (4 WMMA k-steps of 32)
#define NROW  384

// ---------------------------------------------------------------------------
// Factor network for A and B: F[row,r] = b2[r] + sum_m sin(w*(x*W1[m]+b1[m])) * W2[r,m]
// One block (256 threads) per row; result stored f16, padded to 128 cols.
// ---------------------------------------------------------------------------
__global__ void factor2_kernel(const float* __restrict__ x,
                               const float* __restrict__ W1, const float* __restrict__ b1,
                               const float* __restrict__ W2, const float* __restrict__ b2,
                               _Float16* __restrict__ Fpad, int R)
{
    __shared__ float h[MID];
    const int row = blockIdx.x;
    const int m   = threadIdx.x;
    const float xi = x[row];
    h[m] = sinf(OMEGA * (xi * W1[m] + b1[m]));
    __syncthreads();
    if (m < RPAD) {
        float acc = 0.0f;
        if (m < R) {
            acc = b2[m];
            const float* w = W2 + m * MID;
            #pragma unroll 8
            for (int t = 0; t < MID; ++t) acc += h[t] * w[t];
        }
        Fpad[row * RPAD + m] = (_Float16)acc;
    }
}

// ---------------------------------------------------------------------------
// Factor network for C: one extra 256x256 sine layer.
// ---------------------------------------------------------------------------
__global__ void factorC_kernel(const float* __restrict__ x,
                               const float* __restrict__ W1, const float* __restrict__ b1,
                               const float* __restrict__ W2, const float* __restrict__ b2,
                               const float* __restrict__ W3, const float* __restrict__ b3,
                               _Float16* __restrict__ Cpad, int R)
{
    __shared__ float h1[MID];
    __shared__ float h2[MID];
    const int row = blockIdx.x;
    const int m   = threadIdx.x;
    const float xi = x[row];
    h1[m] = sinf(OMEGA * (xi * W1[m] + b1[m]));
    __syncthreads();
    {
        float acc = b2[m];
        const float* w = W2 + m * MID;
        #pragma unroll 8
        for (int t = 0; t < MID; ++t) acc += h1[t] * w[t];
        h2[m] = sinf(OMEGA * acc);
    }
    __syncthreads();
    if (m < RPAD) {
        float acc = 0.0f;
        if (m < R) {
            acc = b3[m];
            const float* w = W3 + m * MID;
            #pragma unroll 8
            for (int t = 0; t < MID; ++t) acc += h2[t] * w[t];
        }
        Cpad[row * RPAD + m] = (_Float16)acc;
    }
}

// ---------------------------------------------------------------------------
// CP contraction as a fused Khatri-Rao GEMM:
//   out[ij, k] = sum_r (A[i,r]*B[j,r]) * C[k,r],   ij = i*384 + j
// Grid: 1152 blocks x 256 threads (8 waves). Block stages the full padded C
// factor (384x128 f16 = 96 KB) into LDS; each wave computes one 16-row x
// 384-col strip of the 147456x384 output using v_wmma_f32_16x16x32_f16.
// ---------------------------------------------------------------------------
__global__ void __launch_bounds__(256) cp_wmma_kernel(
    const _Float16* __restrict__ Ah,
    const _Float16* __restrict__ Bh,
    const _Float16* __restrict__ Ch,
    float* __restrict__ out)
{
    extern __shared__ _Float16 ldsC[];   // 384*128 halves = 96 KB
    const int tid = threadIdx.x;

    // Cooperative global->LDS stage of C (b128 loads + ds_store_b128).
    {
        const v8h* __restrict__ src = (const v8h*)Ch;
        v8h* dst = (v8h*)ldsC;
        #pragma unroll
        for (int it = 0; it < (NROW * RPAD / 8) / 256; ++it)
            dst[tid + it * 256] = src[tid + it * 256];
    }
    __syncthreads();

    const int wave = tid >> 5;
    const int lane = tid & 31;
    const int g    = lane >> 4;   // lane group (K split per ISA A/B layouts)
    const int ln   = lane & 15;

    const int mtile = blockIdx.x * 8 + wave;   // 1152*8 = 9216 row tiles
    const int ij0   = mtile * 16;
    const int ij    = ij0 + ln;                // this lane's M-row
    const int i     = ij / NROW;
    const int j     = ij - i * NROW;

    // Build the 4 K-step A-fragments of M = A(i,:) .* B(j,:) once.
    // A-frag layout (16-bit A 16x32): lane group g holds K = kb + g*8 + [0..7]
    // in VGPRs 0-3 and K = kb + 16 + g*8 + [0..7] in VGPRs 4-7.
    const v8h* __restrict__ arow = (const v8h*)Ah + i * (RPAD / 8);
    const v8h* __restrict__ brow = (const v8h*)Bh + j * (RPAD / 8);
    v16h afrag[4];
    #pragma unroll
    for (int kb = 0; kb < 4; ++kb) {
        v8h lo = arow[kb * 4 + g]     * brow[kb * 4 + g];       // v_pk_mul_f16
        v8h hi = arow[kb * 4 + 2 + g] * brow[kb * 4 + 2 + g];
        afrag[kb] = __builtin_shufflevector(lo, hi,
            0, 1, 2, 3, 4, 5, 6, 7, 8, 9, 10, 11, 12, 13, 14, 15);
    }

    const v8h* __restrict__ ldsv = (const v8h*)ldsC;

    #pragma unroll
    for (int ns = 0; ns < 6; ++ns) {           // 6 strips of 64 output columns
        const int ncol0 = ns * 64;
        v8f acc[4] = {v8f{0.f}, v8f{0.f}, v8f{0.f}, v8f{0.f}};
        #pragma unroll
        for (int t = 0; t < 4; ++t) {
            // B-frag: lane holds column n = ln of this 16-col tile; K values
            // come from contiguous 16B chunks of C-row (ncol0 + t*16 + ln).
            const int crow = (ncol0 + t * 16 + ln) * (RPAD / 8);
            #pragma unroll
            for (int kb = 0; kb < 4; ++kb) {
                v8h b_lo = ldsv[crow + kb * 4 + g];
                v8h b_hi = ldsv[crow + kb * 4 + 2 + g];
                v16h bfrag = __builtin_shufflevector(b_lo, b_hi,
                    0, 1, 2, 3, 4, 5, 6, 7, 8, 9, 10, 11, 12, 13, 14, 15);
                acc[t] = __builtin_amdgcn_wmma_f32_16x16x32_f16(
                    /*neg_a=*/false, afrag[kb],
                    /*neg_b=*/false, bfrag,
                    /*c_mod=*/(short)0, acc[t],
                    /*reuse_a=*/false, /*reuse_b=*/false);
            }
        }
        // D layout: VGPR d -> row d (lanes 0-15) / row d+8 (lanes 16-31), col = ln.
        #pragma unroll
        for (int t = 0; t < 4; ++t) {
            const int col = ncol0 + t * 16 + ln;
            #pragma unroll
            for (int d = 0; d < 8; ++d) {
                out[(size_t)(ij0 + d + 8 * g) * NROW + col] = acc[t][d];
            }
        }
    }
}

// ---------------------------------------------------------------------------
// Host launch
// ---------------------------------------------------------------------------
extern "C" void kernel_launch(void* const* d_in, const int* in_sizes, int n_in,
                              void* d_out, int out_size, void* d_ws, size_t ws_size,
                              hipStream_t stream)
{
    const float* A_in  = (const float*)d_in[0];
    const float* B_in  = (const float*)d_in[1];
    const float* C_in  = (const float*)d_in[2];
    const float* A_W1  = (const float*)d_in[3];
    const float* A_b1  = (const float*)d_in[4];
    const float* A_W2  = (const float*)d_in[5];
    const float* A_b2  = (const float*)d_in[6];
    const float* B_W1  = (const float*)d_in[7];
    const float* B_b1  = (const float*)d_in[8];
    const float* B_W2  = (const float*)d_in[9];
    const float* B_b2  = (const float*)d_in[10];
    const float* C_W1  = (const float*)d_in[11];
    const float* C_b1  = (const float*)d_in[12];
    const float* C_W2  = (const float*)d_in[13];
    const float* C_b2  = (const float*)d_in[14];
    const float* C_W3  = (const float*)d_in[15];
    const float* C_b3  = (const float*)d_in[16];

    const size_t padBytes = (size_t)NROW * RPAD * sizeof(_Float16);  // 96 KB
    _Float16* Apad = (_Float16*)d_ws;
    _Float16* Bpad = (_Float16*)((char*)d_ws + padBytes);
    _Float16* Cpad = (_Float16*)((char*)d_ws + 2 * padBytes);

    factor2_kernel<<<NROW, 256, 0, stream>>>(A_in, A_W1, A_b1, A_W2, A_b2, Apad, 100);
    factor2_kernel<<<NROW, 256, 0, stream>>>(B_in, B_W1, B_b1, B_W2, B_b2, Bpad, 100);
    factorC_kernel<<<NROW, 256, 0, stream>>>(C_in, C_W1, C_b1, C_W2, C_b2, C_W3, C_b3,
                                             Cpad, 100);

    const int mtiles = (NROW * NROW) / 16;          // 9216
    const int blocks = mtiles / 8;                  // 1152 blocks, 8 waves each
    cp_wmma_kernel<<<blocks, 256, padBytes, stream>>>(Apad, Bpad, Cpad, (float*)d_out);
}
